// GCN_9_86354612453997
// MI455X (gfx1250) — matrix operand order repaired and checked by reading, not compiled, not run on hardware
//
#include <hip/hip_runtime.h>
#include <hip/hip_bf16.h>
#include <cstdint>

// ---------------------------------------------------------------------------
// Types for WMMA fragments (CDNA5 / gfx1250, wave32)
// ---------------------------------------------------------------------------
typedef __attribute__((ext_vector_type(16))) __bf16 v16bf;
typedef __attribute__((ext_vector_type(8)))  float  v8f;
typedef __attribute__((ext_vector_type(4)))  unsigned int u32x4;
typedef __attribute__((ext_vector_type(4)))  float  f32x4;

typedef unsigned short ushort_t;

union FragBF {
    v16bf v;
    u32x4 q[2];
};

// round-to-nearest-even f32 -> bf16 (as raw u16)
__device__ __forceinline__ ushort_t f2bf(float f) {
    unsigned int u = __float_as_uint(f);
    unsigned int r = u + 0x7FFFu + ((u >> 16) & 1u);
    return (ushort_t)(r >> 16);
}

// ---------------------------------------------------------------------------
// f32 -> bf16 elementwise conversion (for the input x)
// ---------------------------------------------------------------------------
__global__ void k_f32_to_bf16(const float* __restrict__ in,
                              ushort_t* __restrict__ out, int n) {
    int i = blockIdx.x * blockDim.x + threadIdx.x;
    if (i < n) out[i] = f2bf(in[i]);
}

// ---------------------------------------------------------------------------
// Weight prep: W[fi,fo] f32 row-major -> Wt[foP,fi] bf16 (transposed, padded)
// ---------------------------------------------------------------------------
__global__ void k_weight_prep(const float* __restrict__ W,
                              ushort_t* __restrict__ Wt,
                              int fi, int fo, int foP) {
    int i = blockIdx.x * blockDim.x + threadIdx.x;
    int total = foP * fi;
    if (i >= total) return;
    int col = i / fi;
    int k   = i - col * fi;
    float v = (col < fo) ? W[(size_t)k * fo + col] : 0.0f;
    Wt[(size_t)col * fi + k] = f2bf(v);
}

// ---------------------------------------------------------------------------
// WMMA GEMM: support[N,foP] (f32) = hb[N,fi] (bf16) x Wt[foP,fi]^T (bf16)
// One wave computes a 16 x (16*NSUB) strip. NSUB is compile-time so the
// accumulators stay in fixed (even-aligned) VGPRs and the inner loop is a
// straight-line B-load + v_wmma sequence (no v_movrels register indexing).
// ---------------------------------------------------------------------------
template <int NSUB>
__global__ void k_gemm_wmma(const ushort_t* __restrict__ hb,
                            const ushort_t* __restrict__ Wt,
                            float* __restrict__ support,
                            int M, int fi, int foP) {
    const int lane = threadIdx.x & 31;
    const int wave = (blockIdx.x * blockDim.x + threadIdx.x) >> 5;

    const int tilesM  = M >> 4;                   // M divisible by 16
    const int nChunks = foP / (16 * NSUB);        // exact by construction
    if (wave >= tilesM * nChunks) return;         // wave-uniform exit

    const int mTile   = wave % tilesM;
    const int colBase = (wave / tilesM) * (16 * NSUB);

    const int laneN = lane & 15;   // row within A-tile / col within B,D tiles
    const int half  = lane >> 4;

    v8f acc[NSUB];
#pragma unroll
    for (int t = 0; t < NSUB; ++t)
#pragma unroll
        for (int v = 0; v < 8; ++v) acc[t][v] = 0.0f;

    const ushort_t* aRow = hb + (size_t)((mTile << 4) + laneN) * fi;
    const ushort_t* bBase = Wt + (size_t)(colBase + laneN) * fi + half * 16;

#pragma unroll 1
    for (int kb = 0; kb < fi; kb += 32) {
        // A 16x32 bf16 fragment: lane holds K = kb + half*16 + {0..15}
        // (packed pairs per VGPR).
        FragBF A;
        A.q[0] = *(const u32x4*)(aRow + kb + half * 16);
        A.q[1] = *(const u32x4*)(aRow + kb + half * 16 + 8);

#pragma unroll
        for (int t = 0; t < NSUB; ++t) {
            // B 32x16 bf16 fragment: lane n = laneN, K = kb + half*16 + {0..15}
            const ushort_t* bCol = bBase + (size_t)(t << 4) * fi + kb;
            FragBF B;
            B.q[0] = *(const u32x4*)(bCol);
            B.q[1] = *(const u32x4*)(bCol + 8);

            acc[t] = __builtin_amdgcn_wmma_f32_16x16x32_bf16(
                false, A.v, false, B.v, (short)0, acc[t], false, false);
        }
    }

    // D 16x16 f32: VGPR v, lanes 0-15 -> M=v, lanes 16-31 -> M=8+v; N = laneN
    const int mBase = (mTile << 4) + half * 8;
#pragma unroll
    for (int t = 0; t < NSUB; ++t) {
        const int col = colBase + (t << 4) + laneN;
#pragma unroll
        for (int v = 0; v < 8; ++v)
            support[(size_t)(mBase + v) * foP + col] = acc[t][v];
    }
}

// ---------------------------------------------------------------------------
// agg[n, c] = bias[c] (c < fo) else 0   (folds "+ b" into the scatter target)
// ---------------------------------------------------------------------------
__global__ void k_init_bias(float* __restrict__ agg,
                            const float* __restrict__ b,
                            int N, int fo, int foP) {
    long long i = (long long)blockIdx.x * blockDim.x + threadIdx.x;
    long long total = (long long)N * foP;
    if (i >= total) return;
    int c = (int)(i % foP);
    agg[i] = (c < fo) ? b[c] : 0.0f;
}

// ---------------------------------------------------------------------------
// Edge scatter: for each edge e and channel chunk c..c+3:
//   agg[dst[e], c..c+3] += w[e] * support[src[e], c..c+3]
// grid.y = fo/4 channel chunks; grid.x covers edges (coalesced e per lane).
// ---------------------------------------------------------------------------
__global__ void k_edge(const float* __restrict__ support,
                       const int* __restrict__ src,
                       const int* __restrict__ dst,
                       const float* __restrict__ w,
                       float* __restrict__ agg,
                       int E, int foP) {
    int e = blockIdx.x * blockDim.x + threadIdx.x;
    if (e >= E) return;
    const int c = blockIdx.y << 2;
    const float we = w[e];
    const f32x4 s = *(const f32x4*)(support + (size_t)src[e] * foP + c);
    float* a = agg + (size_t)dst[e] * foP + c;
    atomicAdd(a + 0, we * s.x);
    atomicAdd(a + 1, we * s.y);
    atomicAdd(a + 2, we * s.z);
    atomicAdd(a + 3, we * s.w);
}

// ---------------------------------------------------------------------------
// ReLU + f32 -> bf16 (next layer's activations). Layers 1..8 have fo == foP.
// ---------------------------------------------------------------------------
__global__ void k_relu_bf16(const float* __restrict__ agg,
                            ushort_t* __restrict__ hb, long long n) {
    long long i = (long long)blockIdx.x * blockDim.x + threadIdx.x;
    if (i >= n) return;
    float v = agg[i];
    hb[i] = f2bf(v > 0.0f ? v : 0.0f);
}

// ---------------------------------------------------------------------------
// log_softmax over 40 channels (stride foP=48): one wave32 per node.
// ---------------------------------------------------------------------------
__global__ void k_log_softmax(const float* __restrict__ agg,
                              float* __restrict__ out,
                              int N, int foP, int fo) {
    const int lane = threadIdx.x & 31;
    const int node = (blockIdx.x * blockDim.x + threadIdx.x) >> 5;
    if (node >= N) return;
    const float* row = agg + (size_t)node * foP;

    const float NEG = -3.0e38f;
    float v0 = (lane < fo)      ? row[lane]      : NEG;   // c = lane (0..31)
    float v1 = (lane + 32 < fo) ? row[lane + 32] : NEG;   // c = 32+lane (lanes 0..7)

    float m = fmaxf(v0, v1);
    for (int d = 16; d > 0; d >>= 1) m = fmaxf(m, __shfl_xor(m, d, 32));

    float s = ((lane < fo) ? __expf(v0 - m) : 0.0f) +
              ((lane + 32 < fo) ? __expf(v1 - m) : 0.0f);
    for (int d = 16; d > 0; d >>= 1) s += __shfl_xor(s, d, 32);

    const float ls = __logf(s) + m;
    float* orow = out + (size_t)node * fo;
    orow[lane] = v0 - ls;                         // lanes 0..31 -> c 0..31
    if (lane + 32 < fo) orow[lane + 32] = v1 - ls;
}

// ---------------------------------------------------------------------------
// Host-side orchestration
// ---------------------------------------------------------------------------
static inline size_t align_up(size_t x, size_t a) { return (x + a - 1) & ~(a - 1); }

extern "C" void kernel_launch(void* const* d_in, const int* in_sizes, int n_in,
                              void* d_out, int out_size, void* d_ws, size_t ws_size,
                              hipStream_t stream) {
    static const int DIMS[10] = {512, 256, 256, 256, 256, 256, 128, 128, 64, 40};

    const float* x    = (const float*)d_in[0];
    const int*   src  = (const int*)d_in[1];
    const int*   dst  = (const int*)d_in[2];
    const float* ew   = (const float*)d_in[3];
    float*       out  = (float*)d_out;

    const int N = in_sizes[0] / DIMS[0];
    const int E = in_sizes[1];

    // Workspace partition
    char*  ws     = (char*)d_ws;
    size_t off    = 0;
    ushort_t* hb  = (ushort_t*)(ws + off); off = align_up(off + (size_t)N * 512 * sizeof(ushort_t), 256);
    float* support= (float*)(ws + off);    off = align_up(off + (size_t)N * 256 * sizeof(float), 256);
    float* agg    = (float*)(ws + off);    off = align_up(off + (size_t)N * 256 * sizeof(float), 256);
    ushort_t* Wt  = (ushort_t*)(ws + off); off = align_up(off + (size_t)512 * 256 * sizeof(ushort_t), 256);
    (void)ws_size;

    const int TB = 256;

    // x -> bf16 activations
    {
        long long n = (long long)N * DIMS[0];
        k_f32_to_bf16<<<(unsigned)((n + TB - 1) / TB), TB, 0, stream>>>(x, hb, (int)n);
    }

    for (int l = 0; l < 9; ++l) {
        const int fi  = DIMS[l];
        const int fo  = DIMS[l + 1];
        const int foP = (fo + 15) & ~15;
        const float* W = (const float*)d_in[4 + 2 * l];
        const float* b = (const float*)d_in[5 + 2 * l];

        // W -> Wt (bf16, transposed, zero-padded columns)
        {
            int n = foP * fi;
            k_weight_prep<<<(n + TB - 1) / TB, TB, 0, stream>>>(W, Wt, fi, fo, foP);
        }

        // support = hb @ W (WMMA bf16 -> f32)
        {
            const int tilesM = N >> 4;
            if (foP % 64 == 0) {
                const int nChunks = foP / 64;
                long long thr = (long long)tilesM * nChunks * 32;
                k_gemm_wmma<4><<<(unsigned)((thr + TB - 1) / TB), TB, 0, stream>>>(
                    hb, Wt, support, N, fi, foP);
            } else {  // foP == 48 (final layer): one chunk of 3 sub-tiles
                long long thr = (long long)tilesM * 32;
                k_gemm_wmma<3><<<(unsigned)((thr + TB - 1) / TB), TB, 0, stream>>>(
                    hb, Wt, support, N, fi, foP);
            }
        }

        // agg = broadcast(bias)
        {
            long long n = (long long)N * foP;
            k_init_bias<<<(unsigned)((n + TB - 1) / TB), TB, 0, stream>>>(agg, b, N, fo, foP);
        }

        // agg[dst] += w * support[src]
        {
            dim3 grid((E + TB - 1) / TB, fo >> 2);
            k_edge<<<grid, TB, 0, stream>>>(support, src, dst, ew, agg, E, foP);
        }

        if (l < 8) {
            long long n = (long long)N * fo;   // fo == foP for layers 1..8
            k_relu_bf16<<<(unsigned)((n + TB - 1) / TB), TB, 0, stream>>>(agg, hb, n);
        } else {
            long long thr = (long long)N * 32;
            k_log_softmax<<<(unsigned)((thr + TB - 1) / TB), TB, 0, stream>>>(
                agg, out, N, foP, fo);
        }
    }
}
